// KmeansCodebook_14190571946134
// MI455X (gfx1250) — matrix operand order repaired and checked by reading, not compiled, not run on hardware
//
#include <hip/hip_runtime.h>
#include <hip/hip_bf16.h>

// ---------------------------------------------------------------------------
// KmeansCodebook on gfx1250 (MI455X)
//   distances = ||x||^2 - 2 x.c^T + ||c||^2 ; labels = argmin ; preds = gather
// Split-bf16 GEMM via v_wmma_f32_16x16x32_bf16 (hi*hi + hi*lo + lo*hi).
// ---------------------------------------------------------------------------

typedef __attribute__((ext_vector_type(16))) __bf16 v16bf;
typedef __attribute__((ext_vector_type(8)))  float  v8f;

#define DIM   256
#define KCB   1024

union BF16x16 {
    unsigned short u[16];
    uint4          q[2];
    v16bf          v;
};

// Round-to-nearest-even f32 -> bf16 (bit trick), plus residual term.
__device__ __forceinline__ unsigned short f32_to_bf16_rne(float f) {
    unsigned u = __float_as_uint(f);
    unsigned r = u + 0x7FFFu + ((u >> 16) & 1u);
    return (unsigned short)(r >> 16);
}

__device__ __forceinline__ void split_bf16(float f, unsigned short& hi, unsigned short& lo) {
    unsigned short h = f32_to_bf16_rne(f);
    float fh = __uint_as_float(((unsigned)h) << 16);
    hi = h;
    lo = f32_to_bf16_rne(f - fh);
}

// ---------------------------------------------------------------------------
// Kernel 1: codebook prep.
//   grid = K blocks, 256 threads (= D). Block k, thread d:
//   write cbT_hi/lo[k*D + d] (row-major split-bf16), reduce ||c_k||^2.
// ---------------------------------------------------------------------------
__global__ __launch_bounds__(256) void kmeans_prep_kernel(
    const float* __restrict__ cb,
    unsigned short* __restrict__ cbT_hi,
    unsigned short* __restrict__ cbT_lo,
    float* __restrict__ c_sq)
{
    const int k = blockIdx.x;
    const int d = threadIdx.x;

    float v = cb[(size_t)k * DIM + d];
    unsigned short h, l;
    split_bf16(v, h, l);
    cbT_hi[(size_t)k * DIM + d] = h;
    cbT_lo[(size_t)k * DIM + d] = l;

    float sq = v * v;
#pragma unroll
    for (int m = 16; m >= 1; m >>= 1)
        sq += __shfl_xor(sq, m);

    __shared__ float red[8];
    const int lane = threadIdx.x & 31;
    const int wv   = threadIdx.x >> 5;
    if (lane == 0) red[wv] = sq;
    __syncthreads();
    if (threadIdx.x == 0) {
        float s = 0.f;
#pragma unroll
        for (int i = 0; i < 8; ++i) s += red[i];
        c_sq[k] = s;
    }
}

// ---------------------------------------------------------------------------
// Kernel 2: main GEMM + epilogue.
//   grid = N/128 blocks, 256 threads = 8 waves; wave owns 16 rows of x.
//   A fragments (hi+lo) for all of D held in registers; sweep 64 col tiles.
// ---------------------------------------------------------------------------
__global__ __launch_bounds__(256) void kmeans_main_kernel(
    const float* __restrict__ x,
    const float* __restrict__ cb,
    const unsigned short* __restrict__ cbT_hi,
    const unsigned short* __restrict__ cbT_lo,
    const float* __restrict__ c_sq,
    float* __restrict__ preds,
    int*   __restrict__ labels,
    float* __restrict__ dist)
{
    const int lane   = threadIdx.x & 31;
    const int wave   = threadIdx.x >> 5;
    const int row0   = blockIdx.x * 128 + wave * 16;   // wave's first row
    const int r      = lane & 15;                      // M for A, N for C
    const int hiHalf = lane >> 4;                      // 0 or 1

    // ---- Load A fragments (x rows, split bf16) + accumulate ||x||^2 -------
    // A 16-bit 16x32 layout: half j -> K = (j&7) + 16*(j>>3) + 8*hiHalf.
    v16bf ahi[8], alo[8];
    float sumsq = 0.f;
    const float* px = x + (size_t)(row0 + r) * DIM + 8 * hiHalf;
#pragma unroll
    for (int c = 0; c < 8; ++c) {
        const float4* p0 = (const float4*)(px + 32 * c);        // j = 0..7
        const float4* p1 = (const float4*)(px + 32 * c + 16);   // j = 8..15
        float4 f0 = p0[0], f1 = p0[1], f2 = p1[0], f3 = p1[1];
        float vals[16] = { f0.x, f0.y, f0.z, f0.w,  f1.x, f1.y, f1.z, f1.w,
                           f2.x, f2.y, f2.z, f2.w,  f3.x, f3.y, f3.z, f3.w };
        BF16x16 A, L;
#pragma unroll
        for (int j = 0; j < 16; ++j) {
            sumsq += vals[j] * vals[j];
            split_bf16(vals[j], A.u[j], L.u[j]);
        }
        ahi[c] = A.v;
        alo[c] = L.v;
    }
    // Combine the two half-lane partial sums -> full row sum on both lanes.
    sumsq += __shfl_xor(sumsq, 16);
    // xsq[e] = ||x_row||^2 for row M = e + 8*hiHalf (dynamic shfl -> ds_bpermute)
    float xsq[8];
#pragma unroll
    for (int e = 0; e < 8; ++e)
        xsq[e] = __shfl(sumsq, e + 8 * hiHalf);

    // ---- Sweep all K columns in 16-wide tiles ------------------------------
    float minv[8];
    int   mini[8];
#pragma unroll
    for (int e = 0; e < 8; ++e) { minv[e] = 3.4e38f; mini[e] = 0; }

    for (int tile = 0; tile < KCB / 16; ++tile) {
        const int colb = tile * 16;
        const float cs = c_sq[colb + r];   // this lane's column norm

        // Prefetch next tile's B rows (gfx1250 global_prefetch_b8).
        if (tile + 1 < KCB / 16)
            __builtin_prefetch(cbT_hi + (size_t)lane * KCB + colb + 16, 0, 0);

        v8f acc = {};
#pragma unroll
        for (int c = 0; c < 8; ++c) {
            // B 32x16 bf16: N = lane&15, half j -> K = (j&7) + 16*(j>>3) + 8*hiHalf.
            const size_t boff = (size_t)(colb + r) * DIM + 32 * c + 8 * hiHalf;
            BF16x16 Bh, Bl;
            const uint4* bh = (const uint4*)(cbT_hi + boff);
            const uint4* bl = (const uint4*)(cbT_lo + boff);
            Bh.q[0] = bh[0]; Bh.q[1] = bh[2];
            Bl.q[0] = bl[0]; Bl.q[1] = bl[2];
            // split product: hi*hi + hi*lo + lo*hi  (~fp32 accuracy)
            acc = __builtin_amdgcn_wmma_f32_16x16x32_bf16(
                      false, ahi[c], false, Bh.v, (short)0, acc, false, false);
            acc = __builtin_amdgcn_wmma_f32_16x16x32_bf16(
                      false, ahi[c], false, Bl.v, (short)0, acc, false, false);
            acc = __builtin_amdgcn_wmma_f32_16x16x32_bf16(
                      false, alo[c], false, Bh.v, (short)0, acc, false, false);
        }

        // Epilogue: distances + running argmin.
        // C layout: VGPR e -> M = e + 8*hiHalf, N = lane&15.
#pragma unroll
        for (int e = 0; e < 8; ++e) {
            const float dv  = xsq[e] - 2.0f * acc[e] + cs;
            const int   col = colb + r;
            dist[(size_t)(row0 + e + 8 * hiHalf) * KCB + col] = dv;
            if (dv < minv[e]) { minv[e] = dv; mini[e] = col; }
        }
    }

    // ---- Cross-lane argmin within each 16-lane half (butterfly) -----------
#pragma unroll
    for (int e = 0; e < 8; ++e) {
        float v = minv[e];
        int   i = mini[e];
#pragma unroll
        for (int m = 8; m >= 1; m >>= 1) {
            float ov = __shfl_xor(v, m);
            int   oi = __shfl_xor(i, m);
            if (ov < v || (ov == v && oi < i)) { v = ov; i = oi; }
        }
        minv[e] = v;
        mini[e] = i;   // all 16 lanes of the half now hold the result
    }

    // Labels: lane 0 -> rows row0+0..7, lane 16 -> rows row0+8..15.
    if (r == 0) {
#pragma unroll
        for (int e = 0; e < 8; ++e)
            labels[row0 + e + 8 * hiHalf] = mini[e];
    }

    // Preds gather: wave copies codebook[label] for its 16 rows (coalesced).
#pragma unroll
    for (int m = 0; m < 16; ++m) {
        const int lbl = __shfl(mini[m & 7], (m >> 3) * 16);
        const float4* src = (const float4*)(cb    + (size_t)lbl        * DIM);
        float4*       dst = (float4*)      (preds + (size_t)(row0 + m) * DIM);
        dst[lane]      = src[lane];
        dst[lane + 32] = src[lane + 32];
    }
}

// ---------------------------------------------------------------------------
extern "C" void kernel_launch(void* const* d_in, const int* in_sizes, int n_in,
                              void* d_out, int out_size, void* d_ws, size_t ws_size,
                              hipStream_t stream) {
    const float* x  = (const float*)d_in[0];   // [N, 256]
    const float* cb = (const float*)d_in[1];   // [1024, 256]
    const int N = in_sizes[0] / DIM;           // 131072

    // Workspace: cbT_hi (512KB) | cbT_lo (512KB) | c_sq (4KB)
    unsigned short* cbT_hi = (unsigned short*)d_ws;
    unsigned short* cbT_lo = cbT_hi + (size_t)DIM * KCB;
    float*          c_sq   = (float*)(cbT_lo + (size_t)DIM * KCB);

    // Outputs (concatenated flat, reference return order):
    //   preds [N*256] f32 | class_labels [N] i32 | distances [N*1024] f32
    float* preds  = (float*)d_out;
    int*   labels = (int*)((float*)d_out + (size_t)N * DIM);
    float* dist   = (float*)d_out + (size_t)N * DIM + N;

    kmeans_prep_kernel<<<KCB, 256, 0, stream>>>(cb, cbT_hi, cbT_lo, c_sq);
    kmeans_main_kernel<<<N / 128, 256, 0, stream>>>(x, cb, cbT_hi, cbT_lo, c_sq,
                                                    preds, labels, dist);
}
